// SwinBlock_69544110456848
// MI455X (gfx1250) — matrix operand order repaired and checked by reading, not compiled, not run on hardware
//
#include <hip/hip_runtime.h>
#include <hip/hip_bf16.h>
#include <math.h>

// ---------------------------------------------------------------------------
// Fused Swin block for MI455X (gfx1250, wave32, WMMA 16x16x32 f16).
// One workgroup (256 thr = 8 waves) per 8x8 window; all intermediates in LDS.
// LDS leading dims padded to kill bank conflicts on WMMA fragment loads.
// ---------------------------------------------------------------------------

typedef __attribute__((ext_vector_type(16))) _Float16 v16h;
typedef __attribute__((ext_vector_type(8)))  _Float16 v8h;
typedef __attribute__((ext_vector_type(2)))  _Float16 v2h;
typedef __attribute__((ext_vector_type(8)))  float    v8f;
typedef __attribute__((ext_vector_type(4)))  float    v4f;

#define C_DIM   128
#define L_TOK   64
#define NHEAD   4
#define HD      32
#define HID     512
#define NWIN_X  28            // 224 / 8
#define NBLK    (8 * 28 * 28) // 6272 windows

// padded leading dims (halves / floats) -> conflict-free lane->bank strides
#define LD32  129   // f32 buffers [64][128]
#define LDC   136   // f16, C=128 rows
#define LDQK  264   // f16, q|k 256 cols
#define LDV   72    // f16, vT / attn 64 cols
#define LDSC  65    // f32 scores 64 cols
#define LDH   520   // f16 hidden 512 cols

// LDS arena layout (bytes), regions reused across phases
#define OFF_A 0        // xraw f32 -> yres f32 -> ybuf f32   (33024 B)
#define OFF_B 33056    // xn f16                             (17408 B)
#define OFF_C 50464    // qk f16 -> attn f16                 (36864 B)
#define OFF_D 87328    // vT f16 -> yn f16                   (18432 B)
#define OFF_E 105760   // scores f32 -> hbuf f16             (66560 B)
#define OFF_F 172320   // ctx f16                            (17408 B)
#define ARENA 190464   // 186 KB of the WGP's 320 KB

// ---- WMMA fragment loaders (contiguous K per lane half, same order in A and B) ------------------

// A (16x32 f16): row = m0 + lane%16; halves 0..15 = 16 contiguous K at
// k0 + 16*(lane/16).  Two 16B loads (ld must be mult of 8 halves).
__device__ __forceinline__ v16h load_A(const _Float16* __restrict__ X, int ld,
                                       int m0, int k0, int lane) {
  const _Float16* p = X + (m0 + (lane & 15)) * ld + k0 + ((lane >> 4) << 4);
  union { v16h v; v8h h[2]; } u;
  u.h[0] = *(const v8h*)(p);
  u.h[1] = *(const v8h*)(p + 8);
  return u.v;
}

// B (32x16 f16) from global weights, Bt[n][k] row-major: col = n0 + lane%16;
// halves 0..15 = 16 contiguous K at k0 + 16*(lane/16).  One 32B load.
__device__ __forceinline__ v16h load_B_g(const _Float16* __restrict__ Bt, int ld,
                                         int n0, int k0, int lane) {
  const _Float16* p = Bt + (n0 + (lane & 15)) * ld + k0 + ((lane >> 4) << 4);
  return *(const v16h*)(p);
}

// Same B layout from LDS: two 16B loads so padded ld (mult of 8) stays aligned.
__device__ __forceinline__ v16h load_B_s(const _Float16* __restrict__ Bt, int ld,
                                         int n0, int k0, int lane) {
  const _Float16* p = Bt + (n0 + (lane & 15)) * ld + k0 + ((lane >> 4) << 4);
  union { v16h v; v8h h[2]; } u;
  u.h[0] = *(const v8h*)(p);
  u.h[1] = *(const v8h*)(p + 8);
  return u.v;
}

__device__ __forceinline__ v8f wmma_f16(v16h a, v16h b, v8f c) {
  return __builtin_amdgcn_wmma_f32_16x16x32_f16(false, a, false, b,
                                                (short)0, c, false, false);
}

// ---- fp32 -> fp16 weight conversion ---------------------------------------
__global__ void cvt_f32_to_f16(const float* __restrict__ s,
                               _Float16* __restrict__ d, int n) {
  int i = blockIdx.x * blockDim.x + threadIdx.x;
  if (i < n) d[i] = (_Float16)s[i];
}

// ---- fused Swin block ------------------------------------------------------
__global__ void __launch_bounds__(256)
swin_block_kernel(const float* __restrict__ x,
                  const float* __restrict__ ln1_g, const float* __restrict__ ln1_b,
                  const float* __restrict__ in_b,  const float* __restrict__ out_b,
                  const float* __restrict__ ln2_g, const float* __restrict__ ln2_b,
                  const float* __restrict__ b1,    const float* __restrict__ b2,
                  const _Float16* __restrict__ wqkv, const _Float16* __restrict__ wo,
                  const _Float16* __restrict__ w1h,  const _Float16* __restrict__ w2h,
                  float* __restrict__ out) {
  extern __shared__ __align__(32) char smem[];
  float*    xraw = (float*)(smem + OFF_A);      // [64][LD32]
  float*    yres = (float*)(smem + OFF_A);      // [64][LD32] (xraw dead)
  float*    ybuf = (float*)(smem + OFF_A);      // [64][LD32] (yres dead)
  _Float16* xn   = (_Float16*)(smem + OFF_B);   // [64][LDC]
  _Float16* qk   = (_Float16*)(smem + OFF_C);   // [64][LDQK]
  _Float16* attn = (_Float16*)(smem + OFF_C);   // [4][64][LDV] (qk dead)
  _Float16* vT   = (_Float16*)(smem + OFF_D);   // [128][LDV]
  _Float16* yn   = (_Float16*)(smem + OFF_D);   // [64][LDC] (vT dead)
  float*    sc   = (float*)(smem + OFF_E);      // [4][64][LDSC]
  _Float16* hbuf = (_Float16*)(smem + OFF_E);   // [64][LDH] (sc dead)
  _Float16* ctx  = (_Float16*)(smem + OFF_F);   // [64][LDC]

  __shared__ float mu_s[64];
  __shared__ float rs_s[64];

  const int tid  = threadIdx.x;
  const int lane = tid & 31;
  const int wv   = tid >> 5;

  const int widx = blockIdx.x;
  const int bimg = widx / (NWIN_X * NWIN_X);
  const int wh   = (widx / NWIN_X) % NWIN_X;
  const int ww   = widx % NWIN_X;

  // Warm the 384 KB f16 weight block into cache (global_prefetch_b8).
  for (int j = tid; j < 3072; j += 256)
    __builtin_prefetch((const char*)wqkv + j * 128, 0, 0);

  // ---- Phase 1: gather window [64 tok][128 ch], b128 loads ----------------
  // i -> (ch, 4-token group); 4 consecutive tokens share a pixel row.
  for (int i = tid; i < 2048; i += 256) {
    int ch = i >> 4, m0 = (i & 15) * 4;
    int gh = wh * 8 + (m0 >> 3), gw = ww * 8 + (m0 & 7);
    v4f val = *(const v4f*)&x[(((long)bimg * C_DIM + ch) * 224 + gh) * 224 + gw];
    for (int j = 0; j < 4; ++j) xraw[(m0 + j) * LD32 + ch] = val[j];
  }
  __syncthreads();

  // ---- Phase 2: LayerNorm 1 -> xn (f16) -----------------------------------
  if (tid < 64) {
    float s = 0.f, s2 = 0.f;
    for (int c = 0; c < C_DIM; ++c) {          // ld=129 => conflict-free
      float v = xraw[tid * LD32 + c];
      s += v; s2 += v * v;
    }
    float mu = s * (1.f / C_DIM);
    float var = s2 * (1.f / C_DIM) - mu * mu;
    mu_s[tid] = mu;
    rs_s[tid] = rsqrtf(var + 1e-5f);
  }
  __syncthreads();
  for (int i = tid; i < L_TOK * C_DIM; i += 256) {
    int m = i >> 7, c = i & 127;
    float v = (xraw[m * LD32 + c] - mu_s[m]) * rs_s[m] * ln1_g[c] + ln1_b[c];
    xn[m * LDC + c] = (_Float16)v;
  }
  __syncthreads();

  // ---- Phase 3: QKV = xn @ Wqkv^T + b (96 16x16 tiles) --------------------
  for (int t = wv; t < 96; t += 8) {
    int mt = t & 3, nt = t >> 2;            // nt in [0,24)
    v8f acc = {};
    for (int kk = 0; kk < 4; ++kk) {
      v16h a = load_A(xn, LDC, mt * 16, kk * 32, lane);
      v16h b = load_B_g(wqkv, C_DIM, nt * 16, kk * 32, lane);
      acc = wmma_f16(a, b, acc);
    }
    int n  = nt * 16 + (lane & 15);
    int mb = mt * 16 + ((lane >> 4) << 3);
    float bias = in_b[n];
    if (nt < 8) {             // Q, pre-scaled by 1/sqrt(hd)  (uniform branch)
      for (int i = 0; i < 8; ++i)
        qk[(mb + i) * LDQK + n] = (_Float16)((acc[i] + bias) * 0.17677669529f);
    } else if (nt < 16) {     // K
      for (int i = 0; i < 8; ++i)
        qk[(mb + i) * LDQK + n] = (_Float16)(acc[i] + bias);
    } else {                  // V, stored transposed: vT[d][token]
      for (int i = 0; i < 8; ++i)
        vT[(n - 256) * LDV + (mb + i)] = (_Float16)(acc[i] + bias);
    }
  }
  __syncthreads();

  // ---- Phase 4: scores[h][q][k] = q . k (64 tiles, K=hd=32) ---------------
  for (int t = wv; t < 64; t += 8) {
    int head = t >> 4, qt = (t >> 2) & 3, kt = t & 3;
    v16h a = load_A(qk, LDQK, qt * 16, head * HD, lane);           // Q rows
    v16h b = load_B_s(qk + 128, LDQK, kt * 16, head * HD, lane);   // K rows
    v8f acc = {};
    acc = wmma_f16(a, b, acc);
    int n = kt * 16 + (lane & 15), mb = qt * 16 + ((lane >> 4) << 3);
    float* S = sc + head * 64 * LDSC;
    for (int i = 0; i < 8; ++i) S[(mb + i) * LDSC + n] = acc[i];
  }
  __syncthreads();

  // ---- Phase 5: softmax, 256 threads <-> 256 (head,row) pairs -------------
  {
    int head = tid >> 6, row = tid & 63;
    const float* S = sc + (head * 64 + row) * LDSC;  // ld=65 => conflict-free
    float mx = -1e30f;
    for (int k = 0; k < 64; ++k) mx = fmaxf(mx, S[k]);
    float sum = 0.f;
    for (int k = 0; k < 64; ++k) sum += __expf(S[k] - mx);
    float inv = 1.f / sum;
    _Float16* A = attn + (head * 64 + row) * LDV;
    for (int k = 0; k < 64; k += 2) {
      v2h pk;
      pk[0] = (_Float16)(__expf(S[k] - mx) * inv);
      pk[1] = (_Float16)(__expf(S[k + 1] - mx) * inv);
      *(v2h*)&A[k] = pk;
    }
  }
  __syncthreads();

  // ---- Phase 6: ctx = attn @ V (32 tiles, K=64) ---------------------------
  for (int t = wv; t < 32; t += 8) {
    int head = t >> 3, qt = (t >> 1) & 3, nt = t & 1;
    v8f acc = {};
    for (int ks = 0; ks < 2; ++ks) {
      v16h a = load_A(attn + head * 64 * LDV, LDV, qt * 16, ks * 32, lane);
      v16h b = load_B_s(vT, LDV, head * HD + nt * 16, ks * 32, lane);
      acc = wmma_f16(a, b, acc);
    }
    int n = head * HD + nt * 16 + (lane & 15);
    int mb = qt * 16 + ((lane >> 4) << 3);
    for (int i = 0; i < 8; ++i) ctx[(mb + i) * LDC + n] = (_Float16)acc[i];
  }
  __syncthreads();

  // ---- Phase 7: out-proj + residual (off the NORMED input, per reference) -
  for (int t = wv; t < 32; t += 8) {
    int mt = t & 3, nt = t >> 2;
    v8f acc = {};
    for (int kk = 0; kk < 4; ++kk) {
      v16h a = load_A(ctx, LDC, mt * 16, kk * 32, lane);
      v16h b = load_B_g(wo, C_DIM, nt * 16, kk * 32, lane);
      acc = wmma_f16(a, b, acc);
    }
    int n = nt * 16 + (lane & 15), mb = mt * 16 + ((lane >> 4) << 3);
    float bias = out_b[n];
    for (int i = 0; i < 8; ++i)
      yres[(mb + i) * LD32 + n] =
          acc[i] + bias + (float)xn[(mb + i) * LDC + n];
  }
  __syncthreads();

  // ---- Phase 8: LayerNorm 2 -> yn (f16, reuses vT region) -----------------
  if (tid < 64) {
    float s = 0.f, s2 = 0.f;
    for (int c = 0; c < C_DIM; ++c) {
      float v = yres[tid * LD32 + c];
      s += v; s2 += v * v;
    }
    float mu = s * (1.f / C_DIM);
    float var = s2 * (1.f / C_DIM) - mu * mu;
    mu_s[tid] = mu;
    rs_s[tid] = rsqrtf(var + 1e-5f);
  }
  __syncthreads();
  for (int i = tid; i < L_TOK * C_DIM; i += 256) {
    int m = i >> 7, c = i & 127;
    float v = (yres[m * LD32 + c] - mu_s[m]) * rs_s[m] * ln2_g[c] + ln2_b[c];
    yn[m * LDC + c] = (_Float16)v;
  }
  __syncthreads();

  // ---- Phase 9: MLP1 + exact GELU (128 tiles) -----------------------------
  for (int t = wv; t < 128; t += 8) {
    int mt = t & 3, nt = t >> 2;            // nt in [0,32)
    v8f acc = {};
    for (int kk = 0; kk < 4; ++kk) {
      v16h a = load_A(yn, LDC, mt * 16, kk * 32, lane);
      v16h b = load_B_g(w1h, C_DIM, nt * 16, kk * 32, lane);
      acc = wmma_f16(a, b, acc);
    }
    int n = nt * 16 + (lane & 15), mb = mt * 16 + ((lane >> 4) << 3);
    float bias = b1[n];
    for (int i = 0; i < 8; ++i) {
      float v = acc[i] + bias;
      v = 0.5f * v * (1.f + erff(v * 0.70710678f));
      hbuf[(mb + i) * LDH + n] = (_Float16)v;
    }
  }
  __syncthreads();

  // ---- Phase 10: MLP2 + residual (off normed yn) -> ybuf ------------------
  for (int t = wv; t < 32; t += 8) {
    int mt = t & 3, nt = t >> 2;
    v8f acc = {};
    for (int kk = 0; kk < 16; ++kk) {
      v16h a = load_A(hbuf, LDH, mt * 16, kk * 32, lane);
      v16h b = load_B_g(w2h, HID, nt * 16, kk * 32, lane);
      acc = wmma_f16(a, b, acc);
    }
    int n = nt * 16 + (lane & 15), mb = mt * 16 + ((lane >> 4) << 3);
    float bias = b2[n];
    for (int i = 0; i < 8; ++i) {
      int m = mb + i;
      ybuf[m * LD32 + n] = acc[i] + bias + (float)yn[m * LDC + n];
    }
  }
  __syncthreads();

  // ---- Phase 11: coalesced b128 store to [B,C,H,W] ------------------------
  for (int i = tid; i < 2048; i += 256) {
    int ch = i >> 4, m0 = (i & 15) * 4;
    int gh = wh * 8 + (m0 >> 3), gw = ww * 8 + (m0 & 7);
    v4f val;
    for (int j = 0; j < 4; ++j) val[j] = ybuf[(m0 + j) * LD32 + ch];
    *(v4f*)&out[(((long)bimg * C_DIM + ch) * 224 + gh) * 224 + gw] = val;
  }
}

// ---------------------------------------------------------------------------
extern "C" void kernel_launch(void* const* d_in, const int* in_sizes, int n_in,
                              void* d_out, int out_size, void* d_ws, size_t ws_size,
                              hipStream_t stream) {
  (void)in_sizes; (void)n_in; (void)out_size; (void)ws_size;
  const float* x      = (const float*)d_in[0];
  const float* ln1_g  = (const float*)d_in[1];
  const float* ln1_b  = (const float*)d_in[2];
  const float* inW    = (const float*)d_in[3];   // [384,128]
  const float* inB    = (const float*)d_in[4];
  const float* outW   = (const float*)d_in[5];   // [128,128]
  const float* outB   = (const float*)d_in[6];
  const float* ln2_g  = (const float*)d_in[7];
  const float* ln2_b  = (const float*)d_in[8];
  const float* w1     = (const float*)d_in[9];   // [512,128]
  const float* b1     = (const float*)d_in[10];
  const float* w2     = (const float*)d_in[11];  // [128,512]
  const float* b2     = (const float*)d_in[12];
  float* out          = (float*)d_out;

  // f16 weights in workspace (384 KB, contiguous)
  _Float16* ws = (_Float16*)d_ws;
  _Float16* wqkv = ws;                 // 384*128 = 49152
  _Float16* wo   = ws + 49152;         // 128*128 = 16384
  _Float16* w1h  = ws + 65536;         // 512*128 = 65536
  _Float16* w2h  = ws + 131072;        // 128*512 = 65536

  cvt_f32_to_f16<<<(49152 + 255) / 256, 256, 0, stream>>>(inW,  wqkv, 49152);
  cvt_f32_to_f16<<<(16384 + 255) / 256, 256, 0, stream>>>(outW, wo,   16384);
  cvt_f32_to_f16<<<(65536 + 255) / 256, 256, 0, stream>>>(w1,   w1h,  65536);
  cvt_f32_to_f16<<<(65536 + 255) / 256, 256, 0, stream>>>(w2,   w2h,  65536);

  // allow 186 KB dynamic LDS (gfx1250 WGP has 320 KB)
  hipFuncSetAttribute((const void*)swin_block_kernel,
                      hipFuncAttributeMaxDynamicSharedMemorySize, ARENA);

  swin_block_kernel<<<NBLK, 256, ARENA, stream>>>(
      x, ln1_g, ln1_b, inB, outB, ln2_g, ln2_b, b1, b2,
      wqkv, wo, w1h, w2h, out);
}
